// GNN_81200651698591
// MI455X (gfx1250) — matrix operand order repaired and checked by reading, not compiled, not run on hardware
//
#include <hip/hip_runtime.h>
#include <math.h>

// Problem constants (match reference).
constexpr int kB = 8;
constexpr int kE = 2048;
constexpr int kD = 256;
constexpr int kNE = 262144;
constexpr int kNSTEP = 3;
constexpr int kM = kB * kE;   // 16384 rows (B*E)
constexpr int kG = 4 * kD;    // 1024 gate columns
constexpr int kLDA = kD + 4;  // padded LDS row stride (floats)

typedef __attribute__((ext_vector_type(16))) __bf16 v16bf;
typedef __attribute__((ext_vector_type(8)))  float  v8f;
typedef __attribute__((ext_vector_type(4)))  int    v4i;

#define GLOBAL_AS __attribute__((address_space(1)))
#define LDS_AS __attribute__((address_space(3)))

#if defined(__HIP_DEVICE_COMPILE__) && defined(__gfx1250__) && \
    __has_builtin(__builtin_amdgcn_global_load_async_to_lds_b128)
#define HAS_ASYNC_LDS 1
#endif

__device__ inline v8f zero_v8f() {
  v8f z;
#pragma unroll
  for (int i = 0; i < 8; ++i) z[i] = 0.0f;
  return z;
}

// ---- Stage a 16-row x 256-col fp32 A tile into LDS -------------------------
// 256 threads: thread t stages row t>>4, floats (t&15)*16 .. +15 (4 x b128).
// Uses gfx1250 async global->LDS copies when available.
__device__ inline void stage_a_tile(const float* __restrict__ A, int m0, int K,
                                    float* __restrict__ smem) {
  __syncthreads();  // previous consumers of smem are done
  const int t = threadIdx.x;
  const int row = t >> 4;
  const int fbase = (t & 15) * 16;
  const float* g = A + (size_t)(m0 + row) * K + fbase;
  float* l = smem + row * kLDA + fbase;
#if defined(HAS_ASYNC_LDS)
#pragma unroll
  for (int q = 0; q < 4; ++q) {
    const void* gv = (const void*)(g + q * 4);
    void* lv = (void*)(l + q * 4);
    __builtin_amdgcn_global_load_async_to_lds_b128(
        (GLOBAL_AS v4i*)(GLOBAL_AS const void*)gv,
        (LDS_AS v4i*)(LDS_AS void*)lv, 0, 0);
  }
#if __has_builtin(__builtin_amdgcn_s_wait_asynccnt)
  __builtin_amdgcn_s_wait_asynccnt(0);
#else
  asm volatile("s_wait_asynccnt 0" ::: "memory");
#endif
#else
#pragma unroll
  for (int q = 0; q < 4; ++q)
    *(float4*)(l + q * 4) = *(const float4*)(g + q * 4);
#endif
  __syncthreads();
}

// ---- WMMA bf16 fragment loaders (fp32 source, inline cvt to bf16) ----------
// A fragment from LDS-staged tile: 16x32 (MxK). lanes 0-15: row=lane,
// elems i <- K=k0+i. lanes 16-31: elems i <- K=k0+16+i.
__device__ inline v16bf load_a_frag_lds(const float* __restrict__ smem, int k0,
                                        int lane) {
  const int off = (lane & 16) ? 16 : 0;
  const float* rowp = smem + (lane & 15) * kLDA;
  const float4* p0 = (const float4*)(rowp + k0 + off);
  const float4* p1 = (const float4*)(rowp + k0 + 8 + off);
  v16bf a;
#pragma unroll
  for (int q = 0; q < 2; ++q) {
    float4 f = p0[q];
    a[4 * q + 0] = (__bf16)f.x; a[4 * q + 1] = (__bf16)f.y;
    a[4 * q + 2] = (__bf16)f.z; a[4 * q + 3] = (__bf16)f.w;
  }
#pragma unroll
  for (int q = 0; q < 2; ++q) {
    float4 f = p1[q];
    a[8 + 4 * q + 0] = (__bf16)f.x; a[8 + 4 * q + 1] = (__bf16)f.y;
    a[8 + 4 * q + 2] = (__bf16)f.z; a[8 + 4 * q + 3] = (__bf16)f.w;
  }
  return a;
}

// B fragment: 32x16 (KxN) with B[k][n] = W[n][k], W fp32 row-major [N,K].
// lanes 0-15: elems i <- K=k0+i ; lanes 16-31: elems i <- K=k0+16+i.
__device__ inline v16bf load_b_frag_f32(const float* __restrict__ W, int col,
                                        int K, int k0, int lane) {
  const int off = (lane & 16) ? 16 : 0;
  const float4* p = (const float4*)(W + (size_t)col * K + k0 + off);
  v16bf b;
#pragma unroll
  for (int q = 0; q < 4; ++q) {
    float4 f = p[q];
    b[4 * q + 0] = (__bf16)f.x; b[4 * q + 1] = (__bf16)f.y;
    b[4 * q + 2] = (__bf16)f.z; b[4 * q + 3] = (__bf16)f.w;
  }
  return b;
}

// ---- Block-wide reduction (256 threads) ------------------------------------
__device__ inline float block_sum256(float v, float* red) {
  const int t = threadIdx.x;
  red[t] = v;
  __syncthreads();
#pragma unroll
  for (int s = 128; s > 0; s >>= 1) {
    if (t < s) red[t] += red[t + s];
    __syncthreads();
  }
  float r = red[0];
  __syncthreads();
  return r;
}

// ---- LayerNorm over last dim (256), one row per block ----------------------
__global__ __launch_bounds__(256) void ln_rows_kernel(
    const float* __restrict__ in, const float* __restrict__ g,
    const float* __restrict__ b, float* __restrict__ out) {
  __shared__ float red[256];
  const int m = blockIdx.x, d = threadIdx.x;
  const float x = in[(size_t)m * kD + d];
  const float mu = block_sum256(x, red) * (1.0f / kD);
  const float dx = x - mu;
  const float var = block_sum256(dx * dx, red) * (1.0f / kD);
  out[(size_t)m * kD + d] = dx * rsqrtf(var + 1e-5f) * g[d] + b[d];
}

// ---- init: ent_label = topic_label, loss = 0 -------------------------------
__global__ __launch_bounds__(256) void init_kernel(
    const float* __restrict__ topic, float* __restrict__ elab,
    float* __restrict__ loss) {
  const int i = blockIdx.x * 256 + threadIdx.x;
  if (i < kM) elab[i] = topic[i];
  if (i == 0) *loss = 0.0f;
}

__global__ __launch_bounds__(256) void zero4_kernel(float4* __restrict__ p,
                                                    int n4) {
  const int i = blockIdx.x * 256 + threadIdx.x;
  if (i < n4) p[i] = make_float4(0.f, 0.f, 0.f, 0.f);
}

// ---- Fused rel-linear GEMM + DistMult score + logsigmoid loss --------------
// One block = 16 edges; wave w covers 32 output columns (8 waves * 32 = 256).
__global__ __launch_bounds__(256) void kge_kernel(
    const float* __restrict__ frel,   // [NE, D]
    const float* __restrict__ relW,   // [D, D] (row n = output col n)
    const float* __restrict__ relb,   // [D]
    const float* __restrict__ ent,    // [M, D] layer-normed entities
    const int* __restrict__ hidx, const int* __restrict__ tidx,
    float* __restrict__ loss) {
  __shared__ __align__(16) float smA[16 * kLDA];
  __shared__ float red2[16][8];
  const int lane = threadIdx.x & 31;
  const int wv = threadIdx.x >> 5;
  const int e0 = blockIdx.x * 16;
  const int n0 = wv * 32;
  const int col0 = n0 + (lane & 15);
  const int col1 = col0 + 16;

  stage_a_tile(frel, e0, kD, smA);

  v8f acc0 = zero_v8f();
  v8f acc1 = zero_v8f();
  for (int k0 = 0; k0 < kD; k0 += 32) {
    v16bf a  = load_a_frag_lds(smA, k0, lane);
    v16bf b0 = load_b_frag_f32(relW, col0, kD, k0, lane);
    v16bf b1 = load_b_frag_f32(relW, col1, kD, k0, lane);
    acc0 = __builtin_amdgcn_wmma_f32_16x16x32_bf16(false, a, false, b0,
                                                   (short)0, acc0, false, false);
    acc1 = __builtin_amdgcn_wmma_f32_16x16x32_bf16(false, a, false, b1,
                                                   (short)0, acc1, false, false);
  }

  const int rbase = (lane & 16) ? 8 : 0;
  const float rb0 = relb[col0];
  const float rb1 = relb[col1];
  float part[8];
#pragma unroll
  for (int v = 0; v < 8; ++v) {
    const int e = e0 + rbase + v;
    const int hh = hidx[e];
    const int tt = tidx[e];
    const float ht0 = ent[(size_t)hh * kD + col0] * ent[(size_t)tt * kD + col0];
    const float ht1 = ent[(size_t)hh * kD + col1] * ent[(size_t)tt * kD + col1];
    part[v] = (acc0[v] + rb0) * ht0 + (acc1[v] + rb1) * ht1;
  }
  // Reduce across the 16 lanes of each half-wave (masks < 16 stay in-half).
#pragma unroll
  for (int v = 0; v < 8; ++v) {
#pragma unroll
    for (int msk = 1; msk < 16; msk <<= 1)
      part[v] += __shfl_xor(part[v], msk, 32);
  }
  if ((lane & 15) == 0) {
#pragma unroll
    for (int v = 0; v < 8; ++v) red2[rbase + v][wv] = part[v];
  }
  __syncthreads();
  if (threadIdx.x < 16) {
    float s = 0.0f;
#pragma unroll
    for (int j = 0; j < 8; ++j) s += red2[threadIdx.x][j];
    // -log_sigmoid(s) = softplus(-s)
    const float sp = fmaxf(-s, 0.0f) + log1pf(expf(-fabsf(s)));
    atomicAdd(loss, sp * (1.0f / (float)kNE));
  }
}

// ---- Edge message + scatter (segment_sum via atomics) ----------------------
// 4 edges per block, 64 threads per edge, float4 (b128) loads.
__global__ __launch_bounds__(256) void edge_msg_kernel(
    const float* __restrict__ instr,  // [B, D] (this step)
    const float* __restrict__ frel,   // [NE, D]
    const float* __restrict__ elab,   // [M]
    const int* __restrict__ hidx, const int* __restrict__ tidx,
    const int* __restrict__ bids,
    float* __restrict__ neigh,        // [M, D]
    float* __restrict__ inter) {      // [M]
  const int e = blockIdx.x * 4 + (threadIdx.x >> 6);
  const int d4 = (threadIdx.x & 63) * 4;
  const int b = bids[e];
  const int t = tidx[e];
  const float fp = elab[hidx[e]];
  const float4 qi = *(const float4*)(instr + b * kD + d4);
  const float4 fr = *(const float4*)(frel + (size_t)e * kD + d4);
  float* np = neigh + (size_t)t * kD + d4;
  atomicAdd(np + 0, fp * fmaxf(qi.x * fr.x, 0.0f));
  atomicAdd(np + 1, fp * fmaxf(qi.y * fr.y, 0.0f));
  atomicAdd(np + 2, fp * fmaxf(qi.z * fr.z, 0.0f));
  atomicAdd(np + 3, fp * fmaxf(qi.w * fr.w, 0.0f));
  if ((threadIdx.x & 63) == 0) atomicAdd(&inter[t], fp);
}

// ---- bf16 WMMA GEMM:  C[m,n] = sum_k A1[m,k]W1[n,k] (+ A2[m,k]W2[n,k]) + bias[n]
// block = 256 thr (8 waves); tile 16 rows x 256 cols; wave -> 32 cols (2 accums).
// A tile staged in LDS (async copy); B (weights) streamed from global/L2.
__global__ __launch_bounds__(256) void gemm_dual_bf16_kernel(
    const float* __restrict__ A1, const float* __restrict__ W1,
    const float* __restrict__ A2, const float* __restrict__ W2,
    const float* __restrict__ bias, float* __restrict__ C, int N, int K) {
  __shared__ __align__(16) float smA[16 * kLDA];
  const int lane = threadIdx.x & 31;
  const int wv = threadIdx.x >> 5;
  const int m0 = blockIdx.x * 16;
  const int n0 = blockIdx.y * 256 + wv * 32;
  const int col0 = n0 + (lane & 15);
  const int col1 = col0 + 16;

  v8f acc0 = zero_v8f();
  v8f acc1 = zero_v8f();
  const float* As[2] = {A1, A2};
  const float* Ws[2] = {W1, W2};
#pragma unroll
  for (int p = 0; p < 2; ++p) {
    const float* A = As[p];
    const float* W = Ws[p];
    if (A == nullptr) continue;  // uniform branch: EXEC stays all-ones
    stage_a_tile(A, m0, K, smA);
    for (int k0 = 0; k0 < K; k0 += 32) {
      v16bf a  = load_a_frag_lds(smA, k0, lane);
      v16bf b0 = load_b_frag_f32(W, col0, K, k0, lane);
      v16bf b1 = load_b_frag_f32(W, col1, K, k0, lane);
      acc0 = __builtin_amdgcn_wmma_f32_16x16x32_bf16(false, a, false, b0,
                                                     (short)0, acc0, false, false);
      acc1 = __builtin_amdgcn_wmma_f32_16x16x32_bf16(false, a, false, b1,
                                                     (short)0, acc1, false, false);
    }
  }
  const int rbase = m0 + ((lane & 16) ? 8 : 0);
  const float bz0 = bias[col0];
  const float bz1 = bias[col1];
#pragma unroll
  for (int v = 0; v < 8; ++v) {
    C[(size_t)(rbase + v) * N + col0] = acc0[v] + bz0;
    C[(size_t)(rbase + v) * N + col1] = acc1[v] + bz1;
  }
}

// ---- LSTM elementwise + LN(h) + LN(c) + masked score, one row per block ----
__global__ __launch_bounds__(256) void lstm_kernel(
    const float* __restrict__ gates,   // [M, 4D] order i,f,o,m
    const float* __restrict__ c_prev,  // [M, D]
    const float* __restrict__ inter,   // [M]
    const float* __restrict__ elab,    // [M] previous label
    const float* __restrict__ emask,   // [M]
    const float* __restrict__ ln_g, const float* __restrict__ ln_b,
    const float* __restrict__ score_W, const float* __restrict__ score_b,
    float* __restrict__ h_out, float* __restrict__ c_out,
    float* __restrict__ score_out) {
  __shared__ float red[256];
  const int m = blockIdx.x, d = threadIdx.x;
  const size_t g0 = (size_t)m * kG + d;
  const float gi = gates[g0];
  const float gf = gates[g0 + kD];
  const float go = gates[g0 + 2 * kD];
  const float gm = gates[g0 + 3 * kD];
  const float ig = 1.0f / (1.0f + expf(-gi));
  const float fg = 1.0f / (1.0f + expf(-gf));
  const float og = 1.0f / (1.0f + expf(-go));
  const float mg = tanhf(gm);
  const float cn = fg * c_prev[(size_t)m * kD + d] + ig * mg;
  const float hr = og * tanhf(cn);
  // LN(h)
  const float hmu = block_sum256(hr, red) * (1.0f / kD);
  const float hd = hr - hmu;
  const float hvar = block_sum256(hd * hd, red) * (1.0f / kD);
  const float hn = hd * rsqrtf(hvar + 1e-5f) * ln_g[d] + ln_b[d];
  // LN(c)
  const float cmu = block_sum256(cn, red) * (1.0f / kD);
  const float cd = cn - cmu;
  const float cvar = block_sum256(cd * cd, red) * (1.0f / kD);
  const float cl = cd * rsqrtf(cvar + 1e-5f) * ln_g[d] + ln_b[d];
  h_out[(size_t)m * kD + d] = hn;
  c_out[(size_t)m * kD + d] = cl;
  // score = h . score_W + score_b, then masking
  const float ssum = block_sum256(hn * score_W[d], red);
  if (d == 0) {
    const float sc = ssum + score_b[0];
    const float mk =
        (((inter[m] + elab[m]) > 1e-8f) ? 1.0f : 0.0f) * emask[m];
    score_out[m] = mk * sc + (1.0f - mk) * (-1e20f);
  }
}

// ---- softmax over E=2048 per batch row; writes ent_label and labels output -
__global__ __launch_bounds__(256) void softmax_kernel(
    const float* __restrict__ score, float* __restrict__ elab,
    float* __restrict__ lab_out) {
  __shared__ float red[256];
  const int b = blockIdx.x, t = threadIdx.x;
  float v[kE / 256];
  float vmax = -3.4e38f;
#pragma unroll
  for (int i = 0; i < kE / 256; ++i) {
    v[i] = score[b * kE + t + i * 256];
    vmax = fmaxf(vmax, v[i]);
  }
  red[t] = vmax;
  __syncthreads();
#pragma unroll
  for (int s = 128; s > 0; s >>= 1) {
    if (t < s) red[t] = fmaxf(red[t], red[t + s]);
    __syncthreads();
  }
  const float mx = red[0];
  __syncthreads();
  float lsum = 0.0f;
#pragma unroll
  for (int i = 0; i < kE / 256; ++i) {
    v[i] = expf(v[i] - mx);
    lsum += v[i];
  }
  const float tot = block_sum256(lsum, red);
  const float inv = 1.0f / tot;
#pragma unroll
  for (int i = 0; i < kE / 256; ++i) {
    const float p = v[i] * inv;
    elab[b * kE + t + i * 256] = p;
    lab_out[b * kE + t + i * 256] = p;
  }
}

// ---------------------------------------------------------------------------
extern "C" void kernel_launch(void* const* d_in, const int* in_sizes, int n_in,
                              void* d_out, int out_size, void* d_ws,
                              size_t ws_size, hipStream_t stream) {
  (void)in_sizes; (void)n_in; (void)out_size; (void)ws_size;
  const float* instructions   = (const float*)d_in[0];   // [3,8,256]
  const float* entity_emb     = (const float*)d_in[1];   // [8,2048,256]
  const float* fact_relations = (const float*)d_in[2];   // [NE,256]
  const float* topic_label    = (const float*)d_in[3];   // [8,2048]
  const float* entity_mask    = (const float*)d_in[4];   // [8,2048]
  const float* ln_g           = (const float*)d_in[5];   // [256]
  const float* ln_b           = (const float*)d_in[6];   // [256]
  const float* rel_W          = (const float*)d_in[7];   // [256,256]
  const float* rel_b          = (const float*)d_in[8];   // [256]
  const float* W_ih           = (const float*)d_in[9];   // [1024,256]
  const float* W_hh           = (const float*)d_in[10];  // [1024,256]
  const float* b_hh           = (const float*)d_in[11];  // [1024]
  const float* score_W        = (const float*)d_in[12];  // [1,256]
  const float* score_b        = (const float*)d_in[13];  // [1]
  const float* ffn_W          = (const float*)d_in[14];  // [256,256]
  const float* ffn_b          = (const float*)d_in[15];  // [256]
  const int*   head_idx       = (const int*)d_in[16];    // [NE]
  const int*   tail_idx       = (const int*)d_in[17];    // [NE]
  const int*   batch_ids      = (const int*)d_in[18];    // [NE]

  // Workspace layout (fp32): ~151 MB
  float* ws    = (float*)d_ws;
  float* ent   = ws;                          // kM*kD
  float* h     = ent + (size_t)kM * kD;       // kM*kD
  float* c     = h + (size_t)kM * kD;         // kM*kD
  float* neigh = c + (size_t)kM * kD;         // kM*kD
  float* xln   = neigh + (size_t)kM * kD;     // kM*kD
  float* gates = xln + (size_t)kM * kD;       // kM*kG
  float* inter = gates + (size_t)kM * kG;     // kM
  float* elab  = inter + kM;                  // kM
  float* score = elab + kM;                   // kM

  // Output layout: labels [3,8,2048] ++ out_emb [8,2048,256] ++ kge_loss [1]
  float* out        = (float*)d_out;
  float* labels_out = out;
  float* emb_out    = out + (size_t)kNSTEP * kM;
  float* loss_out   = emb_out + (size_t)kM * kD;

  const dim3 blk(256);
  init_kernel<<<dim3((kM + 255) / 256), blk, 0, stream>>>(topic_label, elab,
                                                          loss_out);
  ln_rows_kernel<<<dim3(kM), blk, 0, stream>>>(entity_emb, ln_g, ln_b, ent);
  kge_kernel<<<dim3(kNE / 16), blk, 0, stream>>>(fact_relations, rel_W, rel_b,
                                                 ent, head_idx, tail_idx,
                                                 loss_out);

  for (int step = 0; step < kNSTEP; ++step) {
    zero4_kernel<<<dim3((kM * kD / 4 + 255) / 256), blk, 0, stream>>>(
        (float4*)neigh, kM * kD / 4);
    zero4_kernel<<<dim3((kM / 4 + 255) / 256), blk, 0, stream>>>(
        (float4*)inter, kM / 4);
    edge_msg_kernel<<<dim3(kNE / 4), blk, 0, stream>>>(
        instructions + (size_t)step * kB * kD, fact_relations, elab, head_idx,
        tail_idx, batch_ids, neigh, inter);
    ln_rows_kernel<<<dim3(kM), blk, 0, stream>>>(neigh, ln_g, ln_b, xln);
    const float* hprev = (step == 0) ? ent : h;
    const float* cprev = (step == 0) ? ent : c;
    gemm_dual_bf16_kernel<<<dim3(kM / 16, kG / 256), blk, 0, stream>>>(
        xln, W_ih, hprev, W_hh, b_hh, gates, kG, kD);
    lstm_kernel<<<dim3(kM), blk, 0, stream>>>(gates, cprev, inter, elab,
                                              entity_mask, ln_g, ln_b, score_W,
                                              score_b, h, c, score);
    softmax_kernel<<<dim3(kB), blk, 0, stream>>>(
        score, elab, labels_out + (size_t)step * kM);
  }
  gemm_dual_bf16_kernel<<<dim3(kM / 16, kD / 256), blk, 0, stream>>>(
      h, ffn_W, nullptr, nullptr, ffn_b, emb_out, kD, kD);
}
